// AttentionLayer_66975720014279
// MI455X (gfx1250) — matrix operand order repaired and checked by reading, not compiled, not run on hardware
//
#include <hip/hip_runtime.h>

typedef __attribute__((ext_vector_type(16))) __bf16 v16bf;
typedef __attribute__((ext_vector_type(8)))  float  v8f;
typedef unsigned short u16;

struct Pack32 { uint4 a, b; };
struct BF4    { u16 h[4]; };               // 8 bytes, 4 packed bf16
static_assert(sizeof(Pack32) == 32, "pack size");
static_assert(sizeof(v16bf) == 32, "v16bf size");
static_assert(sizeof(BF4) == 8, "bf4 size");

constexpr int Bn = 32;
constexpr int Sn = 2048;
constexpr int Hn = 1024;
constexpr int ROWS = 64;          // rows (b,s pairs) per workgroup in the main kernel
constexpr int PAD  = 8;           // LDS row padding (elements) to break bank alignment
constexpr int LDA  = Hn + PAD;    // LDS row stride in bf16 elements
constexpr int SCHUNKS = 16;       // S chunks for the context partial kernel

// round-to-nearest-even f32 -> bf16
__device__ inline u16 f2bf(float x) {
  union { float f; unsigned u; } v; v.f = x;
  unsigned r = v.u + 0x7fffu + ((v.u >> 16) & 1u);
  return (u16)(r >> 16);
}

// tanh(x) = 1 - 2/(exp(2x)+1); single v_exp_f32 + single v_rcp_f32
__device__ inline float tanh_fast(float x) {
  return 1.0f - 2.0f * __builtin_amdgcn_rcpf(__expf(2.0f * x) + 1.0f);
}

// ---------------------------------------------------------------------------
// Kernel 1: W_key [K=H][N=H] f32  ->  bf16 B-fragments in WMMA lane order.
// Chunk index = ((ntg*32 + ks)*32 + lane), 16 bf16 per chunk:
// n = ntg*16 + lane%16, kbase = ks*32 + (lane/16)*16, element e = W_key[kbase+e][n].
// ---------------------------------------------------------------------------
__global__ __launch_bounds__(256) void prep_wkey_kernel(
    const float* __restrict__ Wk, u16* __restrict__ Wf) {
  int idx  = blockIdx.x * 256 + threadIdx.x;      // 64*32*32 = 65536 threads
  int lane = idx & 31;
  int ks   = (idx >> 5) & 31;
  int ntg  = idx >> 10;
  int n     = (ntg << 4) + (lane & 15);
  int kbase = (ks << 5) + ((lane >> 4) << 4);
  alignas(16) u16 tmp[16];
#pragma unroll
  for (int e = 0; e < 16; ++e)
    tmp[e] = f2bf(Wk[(size_t)(kbase + e) * Hn + n]);
  uint4* dst = (uint4*)(Wf + (size_t)idx * 16);
  dst[0] = ((const uint4*)tmp)[0];
  dst[1] = ((const uint4*)tmp)[1];
}

// ---------------------------------------------------------------------------
// Kernel 2: query[b][n] = sum_k dec[b][k] * W_query[k][n]   (f32, tiny)
// ---------------------------------------------------------------------------
__global__ __launch_bounds__(256) void query_kernel(
    const float* __restrict__ dec, const float* __restrict__ Wq,
    float* __restrict__ q) {
  int idx = blockIdx.x * 256 + threadIdx.x;       // 32*1024 threads
  int b = idx >> 10, n = idx & (Hn - 1);
  const float* d = dec + b * Hn;
  float acc = 0.0f;
  for (int k = 0; k < Hn; ++k)
    acc = fmaf(d[k], Wq[(size_t)k * Hn + n], acc);
  q[idx] = acc;
}

// ---------------------------------------------------------------------------
// Kernel 3 (main): fused  scores[b,s] = V . tanh(q[b] + enc[b,s] @ W_key)
// WG = 256 threads (8 waves), 64 rows per WG, wave w owns columns [w*128, w*128+128).
// A tile staged in LDS as bf16 (64 x 1024, padded). keys/energy never hit HBM.
// Deterministic: xor-shuffle butterfly + fixed-order cross-wave sum.
// ---------------------------------------------------------------------------
__global__ __launch_bounds__(256) void bahdanau_scores_kernel(
    const float* __restrict__ enc, const u16* __restrict__ Wf,
    const float* __restrict__ q, const float* __restrict__ Vvec,
    float* __restrict__ scores) {
  extern __shared__ char smem_raw[];
  u16*   As    = (u16*)smem_raw;                                    // ROWS*LDA bf16
  float* swred = (float*)(smem_raw + (size_t)ROWS * LDA * sizeof(u16)); // [8][ROWS]

  const int tid  = threadIdx.x;
  const int lane = tid & 31;
  const int wave = tid >> 5;
  const int lmod = lane & 15;
  const int lhi  = lane >> 4;
  const int loK  = lhi ? 16 : 0;  // contiguous k order used here (same as Wf): lanes 0-15 K=0..15, lanes 16-31 K=16..31

  const int row0 = blockIdx.x * ROWS;   // 64 | 2048 so whole WG is in one batch
  const int b    = row0 / Sn;

  // Stage + convert A tile: vectorized float4 loads (16B/instr) -> 4x bf16 (8B) stores.
  const float4* encBase4 = (const float4*)(enc + (size_t)row0 * Hn);
#pragma unroll 4
  for (int i4 = tid; i4 < ROWS * Hn / 4; i4 += 256) {
    float4 v = encBase4[i4];
    int e0 = i4 * 4;
    int r = e0 >> 10, k = e0 & (Hn - 1);
    BF4 p;
    p.h[0] = f2bf(v.x); p.h[1] = f2bf(v.y);
    p.h[2] = f2bf(v.z); p.h[3] = f2bf(v.w);
    *(BF4*)&As[r * LDA + k] = p;    // 8B aligned: LDA*2 % 8 == 0, k % 4 == 0
  }
  __syncthreads();

  float scorep[4][8];
#pragma unroll
  for (int mt = 0; mt < 4; ++mt)
#pragma unroll
    for (int r = 0; r < 8; ++r) scorep[mt][r] = 0.0f;

  const v8f vzero = {0.f, 0.f, 0.f, 0.f, 0.f, 0.f, 0.f, 0.f};

#pragma unroll 1
  for (int nt = 0; nt < 8; ++nt) {
    const int   ntg = wave * 8 + nt;        // global 16-col tile index
    const int   n   = ntg * 16 + lmod;      // lane's output column (C layout: N = lane%16)
    const float qv  = q[b * Hn + n];
    const float vv  = Vvec[n];

    v8f acc[4];
#pragma unroll
    for (int mt = 0; mt < 4; ++mt) acc[mt] = vzero;

    const Pack32* bptr = (const Pack32*)Wf + (size_t)ntg * 32 * 32 + lane;

#pragma unroll 2
    for (int ks = 0; ks < 32; ++ks) {
      // B fragment: coalesced 32B/lane, L2-resident; WGP-scope prefetch of next step.
      Pack32 bp = bptr[(size_t)ks * 32];
      __builtin_prefetch(bptr + (size_t)(ks + 1) * 32, 0, 3);
      v16bf bfrag = __builtin_bit_cast(v16bf, bp);
      const int k0 = ks * 32;

      // Preload all four A fragments (distinct regs) so the 4 WMMAs can go
      // back-to-back without WAR hazard NOPs against the next ds_loads.
      Pack32 apk[4];
#pragma unroll
      for (int mt = 0; mt < 4; ++mt) {
        const u16* ap = &As[(mt * 16 + lmod) * LDA + k0 + loK];
        apk[mt].a = *(const uint4*)(ap);        // K = k0+loK .. +7
        apk[mt].b = *(const uint4*)(ap + 8);    // K = k0+loK+8 .. +15
      }
#pragma unroll
      for (int mt = 0; mt < 4; ++mt) {
        acc[mt] = __builtin_amdgcn_wmma_f32_16x16x32_bf16(
            false, __builtin_bit_cast(v16bf, apk[mt]), false, bfrag,
            (short)0, acc[mt], false, false);
      }
    }

    // Fused epilogue: partial score = V[n] * tanh(c + q[n]).
#pragma unroll
    for (int mt = 0; mt < 4; ++mt)
#pragma unroll
      for (int r = 0; r < 8; ++r) {
        float e = tanh_fast(acc[mt][r] + qv);
        scorep[mt][r] += e * vv;
      }
  }

  // Deterministic reduction over the 16 N-columns of each tile:
  // xor-butterfly stays within each 16-lane half (masks 1,2,4,8).
#pragma unroll
  for (int mt = 0; mt < 4; ++mt)
#pragma unroll
    for (int r = 0; r < 8; ++r) {
      float v = scorep[mt][r];
      v += __shfl_xor(v, 1);
      v += __shfl_xor(v, 2);
      v += __shfl_xor(v, 4);
      v += __shfl_xor(v, 8);
      scorep[mt][r] = v;
    }
  if (lmod == 0) {
    // lane 0 holds rows mt*16+r, lane 16 holds rows mt*16+8+r
#pragma unroll
    for (int mt = 0; mt < 4; ++mt)
#pragma unroll
      for (int r = 0; r < 8; ++r)
        swred[wave * ROWS + mt * 16 + lhi * 8 + r] = scorep[mt][r];
  }
  __syncthreads();

  if (tid < ROWS) {
    float sum = 0.0f;
#pragma unroll
    for (int w2 = 0; w2 < 8; ++w2) sum += swred[w2 * ROWS + tid];  // fixed order
    scores[row0 + tid] = sum;
  }
}

// ---------------------------------------------------------------------------
// Kernel 4: softmax over S per batch (fixed-order tree reductions)
// ---------------------------------------------------------------------------
__global__ __launch_bounds__(256) void softmax_kernel(
    const float* __restrict__ scores, float* __restrict__ attn) {
  __shared__ float red[256];
  const int b = blockIdx.x, tid = threadIdx.x;
  const float* sc = scores + (size_t)b * Sn;
  float* w = attn + (size_t)b * Sn;

  float mx = -3.4e38f;
  for (int s = tid; s < Sn; s += 256) mx = fmaxf(mx, sc[s]);
  red[tid] = mx; __syncthreads();
  for (int off = 128; off > 0; off >>= 1) {
    if (tid < off) red[tid] = fmaxf(red[tid], red[tid + off]);
    __syncthreads();
  }
  mx = red[0]; __syncthreads();

  float sum = 0.0f;
  for (int s = tid; s < Sn; s += 256) {
    float e = __expf(sc[s] - mx);
    w[s] = e;
    sum += e;
  }
  red[tid] = sum; __syncthreads();
  for (int off = 128; off > 0; off >>= 1) {
    if (tid < off) red[tid] += red[tid + off];
    __syncthreads();
  }
  const float inv = 1.0f / red[0];
  for (int s = tid; s < Sn; s += 256) w[s] *= inv;
}

// ---------------------------------------------------------------------------
// Kernel 5a: context partials over S chunks (deterministic, float4 loads)
// part[b][sc][h] = sum_{s in chunk sc} attn[b][s] * enc[b][s][h]
// ---------------------------------------------------------------------------
__global__ __launch_bounds__(256) void context_part_kernel(
    const float* __restrict__ enc, const float* __restrict__ attn,
    float* __restrict__ part) {
  const int b  = blockIdx.x;
  const int sc = blockIdx.y;
  const int h4 = threadIdx.x * 4;
  constexpr int SL = Sn / SCHUNKS;   // 128

  const float*  w = attn + (size_t)b * Sn + sc * SL;
  const float4* e = (const float4*)(enc + (size_t)b * Sn * Hn
                                        + (size_t)sc * SL * Hn + h4);
  float4 acc = {0.f, 0.f, 0.f, 0.f};
  for (int s = 0; s < SL; ++s) {
    const float  ws = w[s];
    const float4 ev = e[(size_t)s * (Hn / 4)];
    acc.x = fmaf(ws, ev.x, acc.x);
    acc.y = fmaf(ws, ev.y, acc.y);
    acc.z = fmaf(ws, ev.z, acc.z);
    acc.w = fmaf(ws, ev.w, acc.w);
  }
  *(float4*)(part + ((size_t)(b * SCHUNKS + sc)) * Hn + h4) = acc;
}

// Kernel 5b: fixed-order reduce of the 16 chunk partials
__global__ __launch_bounds__(256) void context_reduce_kernel(
    const float* __restrict__ part, float* __restrict__ ctx) {
  int idx = blockIdx.x * 256 + threadIdx.x;   // 32*1024
  int b = idx >> 10, h = idx & (Hn - 1);
  float s = 0.0f;
#pragma unroll
  for (int c = 0; c < SCHUNKS; ++c)
    s += part[((size_t)(b * SCHUNKS + c)) * Hn + h];
  ctx[idx] = s;
}

// ---------------------------------------------------------------------------
extern "C" void kernel_launch(void* const* d_in, const int* in_sizes, int n_in,
                              void* d_out, int out_size, void* d_ws, size_t ws_size,
                              hipStream_t stream) {
  const float* enc = (const float*)d_in[0];   // [B,S,H]
  const float* dec = (const float*)d_in[1];   // [B,1,H]
  const float* Wq  = (const float*)d_in[2];   // [H,H]
  const float* Wk  = (const float*)d_in[3];   // [H,H]
  const float* V   = (const float*)d_in[4];   // [H]

  float* out  = (float*)d_out;
  float* ctx  = out;               // [B,H]  (first in tuple order)
  float* attn = out + Bn * Hn;     // [B,S]

  char* ws = (char*)d_ws;
  size_t off = 0;
  u16*   Wf     = (u16*)(ws + off);  off += (size_t)Hn * Hn * sizeof(u16);       // 2 MB
  float* q      = (float*)(ws + off); off += (size_t)Bn * Hn * sizeof(float);    // 128 KB
  float* scores = (float*)(ws + off); off += (size_t)Bn * Sn * sizeof(float);    // 256 KB
  float* part   = (float*)(ws + off); off += (size_t)Bn * SCHUNKS * Hn * sizeof(float); // 2 MB

  prep_wkey_kernel<<<(Hn * Hn / 16) / 256, 256, 0, stream>>>(Wk, Wf);
  query_kernel<<<(Bn * Hn) / 256, 256, 0, stream>>>(dec, Wq, q);

  const size_t smem = (size_t)ROWS * LDA * sizeof(u16)
                    + (size_t)8 * ROWS * sizeof(float);   // ~134 KB (< 320 KB/WGP)
  bahdanau_scores_kernel<<<(Bn * Sn) / ROWS, 256, smem, stream>>>(enc, Wf, q, V, scores);

  softmax_kernel<<<Bn, 256, 0, stream>>>(scores, attn);
  context_part_kernel<<<dim3(Bn, SCHUNKS), 256, 0, stream>>>(enc, attn, part);
  context_reduce_kernel<<<(Bn * Hn) / 256, 256, 0, stream>>>(part, ctx);
}
